// Rits_20151986553217
// MI455X (gfx1250) — matrix operand order repaired and checked by reading, not compiled, not run on hardware
//
#include <hip/hip_runtime.h>

#define RT 36
#define RF 9
#define RH 64

typedef __attribute__((ext_vector_type(16))) _Float16 v16h;
typedef __attribute__((ext_vector_type(8)))  float    v8f;
typedef __attribute__((ext_vector_type(4)))  unsigned int u32x4;

union FragH { v16h v; u32x4 q[2]; };
union FragF { v8f  v; float f[8]; };

#define LOG2E 1.4426950408889634f

// A fragment (16x32 f16): lane l holds row (l&15); halves 0..7 = K kb..kb+7,
// halves 8..15 = K kb+16..kb+23, kb = 8*(l>>4).  Two 16B LDS loads.
__device__ __forceinline__ v16h load_a_frag(const _Float16* base, int kpad, int lane, int ko) {
  const _Float16* p = base + (lane & 15) * kpad + (((lane >> 4) << 3) + ko);
  FragH f;
  f.q[0] = *(const u32x4*)(const void*)(p);
  f.q[1] = *(const u32x4*)(const void*)(p + 16);
  return f.v;
}

// B fragment (32x16 f16), B = W^T with W stored row-major [N][kpad]:
// lane l reads W row (ntile*16 + (l&15)): halves 0..7 = K kb..kb+7,
// halves 8..15 = K kb+16..kb+23, kb = 8*(l>>4)+ko.  Two 16B LDS loads.
__device__ __forceinline__ v16h load_b_frag(const _Float16* base, int kpad, int lane, int ntile, int ko) {
  const _Float16* p = base + ((ntile << 4) + (lane & 15)) * kpad + (((lane >> 4) << 3) + ko);
  FragH f;
  f.q[0] = *(const u32x4*)(const void*)(p);
  f.q[1] = *(const u32x4*)(const void*)(p + 16);
  return f.v;
}

__device__ __forceinline__ v8f wmma16(v16h a, v16h b, v8f c) {
  return __builtin_amdgcn_wmma_f32_16x16x32_f16(false, a, false, b, (short)0, c, false, false);
}

// native v_tanh_f32 on CDNA5 (TRANS32 op)
__device__ __forceinline__ float tanh_fast(float x) {
#if __has_builtin(__builtin_amdgcn_tanhf)
  return __builtin_amdgcn_tanhf(x);
#else
  float r;
  asm("v_tanh_f32 %0, %1\n\tv_nop" : "=v"(r) : "v"(x));
  return r;
#endif
}

__device__ __forceinline__ float exp2_fast(float x) {
#if __has_builtin(__builtin_amdgcn_exp2f)
  return __builtin_amdgcn_exp2f(x);
#else
  return exp2f(x);
#endif
}

__device__ __forceinline__ float sigm(float x) {
  return fmaf(0.5f, tanh_fast(0.5f * x), 0.5f);
}

__global__ void rits_zero(float* p, int n) {
  int i = threadIdx.x + blockIdx.x * blockDim.x;
  if (i < n) p[i] = 0.0f;
}

__global__ void rits_final(const float* __restrict__ num, const float* __restrict__ den,
                           float* __restrict__ loss_out) {
  __shared__ float s[64];
  int t = threadIdx.x;
  float v = 0.0f;
  if (t < RT) v = num[t] / (den[t] + 1e-5f);
  s[t] = v;
  __syncthreads();
  for (int off = 32; off; off >>= 1) {
    if (t < off) s[t] += s[t + off];
    __syncthreads();
  }
  if (t == 0) loss_out[0] = s[0] / (float)RT;
}

__global__ __launch_bounds__(256, 1)
void rits_main(const float* __restrict__ values, const float* __restrict__ masks,
               const float* __restrict__ deltas,
               const float* __restrict__ W_gh, const float* __restrict__ b_gh,
               const float* __restrict__ W_gx, const float* __restrict__ b_gx,
               const float* __restrict__ W_hist, const float* __restrict__ b_hist,
               const float* __restrict__ W_feat, const float* __restrict__ b_feat,
               const float* __restrict__ W_comb, const float* __restrict__ b_comb,
               const float* __restrict__ W_ih, const float* __restrict__ W_hh,
               const float* __restrict__ b_ih, const float* __restrict__ b_hh,
               float* __restrict__ out_imp, float* __restrict__ ws_num,
               float* __restrict__ ws_den)
{
  extern __shared__ __align__(16) char smem_raw[];
  _Float16* Wbig = (_Float16*)smem_raw;        // [256][96]: 0..17 W_ih, 18 bias, 19..31 zero, 32..95 W_hh
  _Float16* Wghl = Wbig + 256 * 96;            // [64][32]:  0..8 W_gh*log2e, 9 b_gh*log2e, rest 0
  _Float16* Whil = Wghl + 64 * 32;             // [16][64]
  _Float16* Wfel = Whil + 16 * 64;             // [16][32]:  0..8 off-diag W_feat, 9 b_feat, rest 0
  _Float16* Wcol = Wfel + 16 * 32;             // [16][32]:  0..17 W_comb, 18 b_comb, rest 0
  float*    fl   = (float*)(Wcol + 16 * 32);
  float* Lbhi = fl;           // 16
  float* Lwgx = fl + 16;      // 16 (scaled by log2e)
  float* Lbgx = fl + 32;      // 16 (scaled by log2e) -> 48 floats
  _Float16* pw = (_Float16*)(fl + 48);         // per-wave staging: 8 * 3072 halves

  const int tid = threadIdx.x;

  // ---- stage weights into LDS as f16 (once per block), biases folded into K columns ----
  {
    int nr = tid; // 256 rows, one per thread
    for (int k = 0; k < 18; ++k) Wbig[nr * 96 + k] = (_Float16)W_ih[nr * 18 + k];
    Wbig[nr * 96 + 18] = (_Float16)(b_ih[nr] + b_hh[nr]);
    for (int k = 19; k < 32; ++k) Wbig[nr * 96 + k] = (_Float16)0.0f;
    for (int k = 0; k < 64; ++k) Wbig[nr * 96 + 32 + k] = (_Float16)W_hh[nr * 64 + k];
  }
  if (tid < 64) {
    for (int k = 0; k < 32; ++k) {
      float w = 0.0f;
      if (k < RF)       w = W_gh[tid * RF + k] * LOG2E;
      else if (k == RF) w = b_gh[tid] * LOG2E;
      Wghl[tid * 32 + k] = (_Float16)w;
    }
  }
  if (tid < 16) {
    for (int k = 0; k < 64; ++k)
      Whil[tid * 64 + k] = (tid < RF) ? (_Float16)W_hist[tid * RH + k] : (_Float16)0.0f;
    for (int k = 0; k < 32; ++k) {
      float wf = 0.0f;
      if (tid < RF) {
        if (k < RF && k != tid) wf = W_feat[tid * RF + k];
        else if (k == RF)       wf = b_feat[tid];
      }
      Wfel[tid * 32 + k] = (_Float16)wf;
      float wc = 0.0f;
      if (tid < RF) {
        if (k < 18)       wc = W_comb[tid * 18 + k];
        else if (k == 18) wc = b_comb[tid];
      }
      Wcol[tid * 32 + k] = (_Float16)wc;
    }
    Lbhi[tid] = (tid < RF) ? b_hist[tid] : 0.0f;
    Lwgx[tid] = (tid < RF) ? W_gx[tid * RF + tid] * LOG2E : 0.0f;
    Lbgx[tid] = (tid < RF) ? b_gx[tid] * LOG2E : 0.0f;
  }
  for (int i = tid; i < 8 * 3072; i += 256) pw[i] = (_Float16)0.0f;
  __syncthreads();
  // constant-1.0 "bias" columns in the staging buffers (never overwritten)
  if (tid < 128) {
    _Float16* base = pw + (tid >> 4) * 3072;
    int row = tid & 15;
    base[1024 + row * 32 + 9]  = (_Float16)1.0f;  // dS  col 9
    base[1536 + row * 32 + 9]  = (_Float16)1.0f;  // xcS col 9
    base[2048 + row * 32 + 18] = (_Float16)1.0f;  // amS col 18
    base[2560 + row * 32 + 18] = (_Float16)1.0f;  // inS col 18
  }
  __syncthreads();

  const int wave = tid >> 5;
  const int lane = tid & 31;
  _Float16* hL  = pw + wave * 3072;   // [16][64] f16 decayed-h staging
  _Float16* dS  = hL + 1024;          // [16][32] deltas (col 9 = 1, 10..31 zero)
  _Float16* xcS = dS + 512;           // [16][32] x_c    (col 9 = 1, 10..31 zero)
  _Float16* amS = xcS + 512;          // [16][32] [gamma_x | m | 1] (19..31 zero)
  _Float16* inS = amS + 512;          // [16][32] [c_c | m | 1]     (19..31 zero)

  const int  n    = lane & 15;         // C-layout column owned by this lane
  const int  roff = (lane >> 4) << 3;  // C-layout row offset (0 or 8)
  const long b0   = (long)blockIdx.x * 128 + (long)wave * 16;
  const bool nv   = (n < RF);

  FragF h[4], c[4];
  #pragma unroll
  for (int j = 0; j < 4; ++j)
    #pragma unroll
    for (int i = 0; i < 8; ++i) { h[j].f[i] = 0.0f; c[j].f[i] = 0.0f; }

  const v8f zero8 = {};

  for (int t = 0; t < RT; ++t) {
    // ---- load x, m, d in C layout (row roff+r, col n) ----
    float xv[8], mv[8], dv[8];
    #pragma unroll
    for (int r = 0; r < 8; ++r) {
      if (nv) {
        long idx = ((b0 + roff + r) * RT + t) * RF + n;
        xv[r] = values[idx]; mv[r] = masks[idx]; dv[r] = deltas[idx];
      } else { xv[r] = 0.0f; mv[r] = 0.0f; dv[r] = 0.0f; }
    }
    if (nv) {
      #pragma unroll
      for (int r = 0; r < 8; ++r) dS[(roff + r) * 32 + n] = (_Float16)dv[r];
    }
    // gamma_x (diagonal temporal decay), log2e prefolded -> bare v_exp
    float gx[8];
    #pragma unroll
    for (int r = 0; r < 8; ++r)
      gx[r] = exp2_fast(-fmaxf(dv[r] * Lwgx[n] + Lbgx[n], 0.0f));

    // ---- gamma_h = exp2(-relu(d@(W_gh*log2e)^T + b*log2e)) ; h *= gamma_h ----
    v16h aD = load_a_frag(dS, 32, lane, 0);
    #pragma unroll
    for (int jt = 0; jt < 4; ++jt) {
      v16h bG = load_b_frag(Wghl, 32, lane, jt, 0);
      FragF g;
      g.v = wmma16(aD, bG, zero8);           // bias folded, C = inline 0
      #pragma unroll
      for (int i = 0; i < 8; ++i) h[jt].f[i] *= exp2_fast(-fmaxf(g.f[i], 0.0f));
    }
    // decayed h -> LDS (f16) for A-side use
    #pragma unroll
    for (int jt = 0; jt < 4; ++jt)
      #pragma unroll
      for (int i = 0; i < 8; ++i)
        hL[(roff + i) * 64 + jt * 16 + n] = (_Float16)h[jt].f[i];

    v16h aH0 = load_a_frag(hL, 64, lane, 0);
    v16h aH1 = load_a_frag(hL, 64, lane, 32);

    // ---- x_h = h @ W_hist^T + b ----
    FragF xh; { float bb = Lbhi[n];
      #pragma unroll
      for (int i = 0; i < 8; ++i) xh.f[i] = bb; }
    xh.v = wmma16(aH0, load_b_frag(Whil, 64, lane, 0, 0),  xh.v);
    xh.v = wmma16(aH1, load_b_frag(Whil, 64, lane, 0, 32), xh.v);

    // ---- x_c ; stage x_c and [gamma_x|m] ----
    float xc[8];
    #pragma unroll
    for (int r = 0; r < 8; ++r) xc[r] = mv[r] * xv[r] + (1.0f - mv[r]) * xh.f[r];
    if (nv) {
      #pragma unroll
      for (int r = 0; r < 8; ++r) {
        xcS[(roff + r) * 32 + n]     = (_Float16)xc[r];
        amS[(roff + r) * 32 + n]     = (_Float16)gx[r];
        amS[(roff + r) * 32 + 9 + n] = (_Float16)mv[r];
      }
    }

    // ---- z_h, alpha (biases folded via constant-1 column) ----
    FragF zh;
    zh.v = wmma16(load_a_frag(xcS, 32, lane, 0), load_b_frag(Wfel, 32, lane, 0, 0), zero8);
    FragF al;
    al.v = wmma16(load_a_frag(amS, 32, lane, 0), load_b_frag(Wcol, 32, lane, 0, 0), zero8);

    // ---- c_h, loss partials, c_c, output, stage [c_c|m] ----
    float lsum = 0.0f, msum = 0.0f;
    #pragma unroll
    for (int r = 0; r < 8; ++r) {
      float ch = al.f[r] * zh.f[r] + (1.0f - al.f[r]) * xh.f[r];
      float cc = mv[r] * xv[r] + (1.0f - mv[r]) * ch;
      if (nv) {
        lsum += (fabsf(xv[r] - xh.f[r]) + fabsf(xv[r] - zh.f[r]) + fabsf(xv[r] - ch)) * mv[r];
        msum += mv[r];
        out_imp[((b0 + roff + r) * RT + t) * RF + n] = cc;
        inS[(roff + r) * 32 + n]     = (_Float16)cc;
        inS[(roff + r) * 32 + 9 + n] = (_Float16)mv[r];
      }
    }
    #pragma unroll
    for (int off = 16; off; off >>= 1) {
      lsum += __shfl_xor(lsum, off, 32);
      msum += __shfl_xor(msum, off, 32);
    }
    if (lane == 0) {
      atomicAdd(&ws_num[t], lsum);
      atomicAdd(&ws_den[t], msum);
    }

    // ---- LSTM gates: [c_c|m|1|pad | h64] (K=96) x 256, bias in K-col 18 ----
    v16h aI = load_a_frag(inS, 32, lane, 0);
    #pragma unroll
    for (int hb = 0; hb < 4; ++hb) {          // 16-wide hidden block
      FragF gq[4];
      #pragma unroll
      for (int q = 0; q < 4; ++q) {           // q: 0=i, 1=f, 2=g, 3=o
        int jt = q * 4 + hb;                  // N-tile index in [0,16)
        FragF g;
        g.v = wmma16(aI,  load_b_frag(Wbig, 96, lane, jt, 0),  zero8);
        g.v = wmma16(aH0, load_b_frag(Wbig, 96, lane, jt, 32), g.v);
        g.v = wmma16(aH1, load_b_frag(Wbig, 96, lane, jt, 64), g.v);
        gq[q] = g;
      }
      #pragma unroll
      for (int i = 0; i < 8; ++i) {
        float ig = sigm(gq[0].f[i]);
        float fg = sigm(gq[1].f[i]);
        float gg = tanh_fast(gq[2].f[i]);
        float og = sigm(gq[3].f[i]);
        float cn = fg * c[hb].f[i] + ig * gg;
        c[hb].f[i] = cn;
        h[hb].f[i] = og * tanh_fast(cn);
      }
    }
  }
}

extern "C" void kernel_launch(void* const* d_in, const int* in_sizes, int n_in,
                              void* d_out, int out_size, void* d_ws, size_t ws_size,
                              hipStream_t stream) {
  const float* values = (const float*)d_in[0];
  const float* masks  = (const float*)d_in[1];
  const float* deltas = (const float*)d_in[2];
  const float* W_gh   = (const float*)d_in[3];
  const float* b_gh   = (const float*)d_in[4];
  const float* W_gx   = (const float*)d_in[5];
  const float* b_gx   = (const float*)d_in[6];
  const float* W_hist = (const float*)d_in[7];
  const float* b_hist = (const float*)d_in[8];
  const float* W_feat = (const float*)d_in[9];
  const float* b_feat = (const float*)d_in[10];
  const float* W_comb = (const float*)d_in[11];
  const float* b_comb = (const float*)d_in[12];
  const float* W_ih   = (const float*)d_in[13];
  const float* W_hh   = (const float*)d_in[14];
  const float* b_ih   = (const float*)d_in[15];
  const float* b_hh   = (const float*)d_in[16];

  float* out    = (float*)d_out;      // out[0] = loss, out[1..] = imputations [B,T,F]
  float* wsf    = (float*)d_ws;
  float* ws_num = wsf;
  float* ws_den = wsf + RT;

  int B = in_sizes[0] / (RT * RF);    // 16384
  int blocks = B / 128;               // 16 rows/wave * 8 waves

  // LDS: weight panels (f16) + small consts (f32) + per-wave staging (f16)
  size_t shmem = (size_t)((256 * 96 + 64 * 32 + 16 * 64 + 16 * 32 + 16 * 32) * 2
                          + 48 * 4 + 8 * 3072 * 2); // = 106688 B

  rits_zero<<<1, 128, 0, stream>>>(wsf, 2 * RT);
  rits_main<<<blocks, 256, shmem, stream>>>(values, masks, deltas,
                                            W_gh, b_gh, W_gx, b_gx,
                                            W_hist, b_hist, W_feat, b_feat,
                                            W_comb, b_comb, W_ih, W_hh, b_ih, b_hh,
                                            out + 1, ws_num, ws_den);
  rits_final<<<1, 64, 0, stream>>>(ws_num, ws_den, out);
}
